// SGS_10247791968409
// MI455X (gfx1250) — matrix operand order repaired and checked by reading, not compiled, not run on hardware
//
#include <hip/hip_runtime.h>

// ---------------- problem constants ----------------
#define BB 4
#define CC 256
#define NN 4096        // H*W
#define GG 8
#define NH 8
#define HD 32
#define MM 512         // N / G
#define HEADSZ (MM*HD) // 16384 elements per (b,g,h)

typedef __attribute__((ext_vector_type(16))) __bf16 v16bf;
typedef __attribute__((ext_vector_type(8)))  __bf16 v8bf;
typedef __attribute__((ext_vector_type(8)))  float  v8f;

static __device__ __forceinline__ __bf16 f2bf(float f) {
    unsigned u = __builtin_bit_cast(unsigned, f);
    u += 0x7FFFu + ((u >> 16) & 1u);              // round-to-nearest-even
    unsigned short h = (unsigned short)(u >> 16);
    return __builtin_bit_cast(__bf16, h);
}

// A-fragment (16x32 bf16): lane L holds row (L%16), K = 16*(L/16)..+15 contiguous
static __device__ __forceinline__ v16bf load_a_frag(const __bf16* __restrict__ rowptr, int k0, int hi) {
    v8bf lo8 = *(const v8bf*)(rowptr + k0 + 16 * hi);
    v8bf hi8 = *(const v8bf*)(rowptr + k0 + 16 * hi + 8);
    return __builtin_shufflevector(lo8, hi8, 0,1,2,3,4,5,6,7,8,9,10,11,12,13,14,15);
}
// B-fragment (32x16 bf16): lane L holds col (L%16), K = 16*(L/16)..+15 contiguous (Bt row-major in K)
static __device__ __forceinline__ v16bf load_b_frag(const __bf16* __restrict__ colptr, int k0, int hi) {
    return *(const v16bf*)(colptr + k0 + 16 * hi);
}

#define WMMA_BF16(A, B, C) \
    __builtin_amdgcn_wmma_f32_16x16x32_bf16(false, (A), false, (B), (short)0, (C), false, false)

// ---------------- kernel 1: pack x [B,C,N] f32 -> xf [B,N,C] bf16 (LDS tiled transpose) ----------------
__global__ __launch_bounds__(256) void pack_x_kernel(const float* __restrict__ x, __bf16* __restrict__ xf) {
    __shared__ float tile[32][33];
    int b  = blockIdx.z;
    int c0 = blockIdx.y * 32;
    int n0 = blockIdx.x * 32;
    int tx = threadIdx.x, ty = threadIdx.y;   // 32 x 8
    const float* xp = x + (size_t)b * CC * NN;
    #pragma unroll
    for (int i = 0; i < 4; ++i)
        tile[ty + 8 * i][tx] = xp[(size_t)(c0 + ty + 8 * i) * NN + n0 + tx];
    __syncthreads();
    __bf16* op = xf + (size_t)b * NN * CC;
    #pragma unroll
    for (int i = 0; i < 4; ++i)
        op[(size_t)(n0 + ty + 8 * i) * CC + c0 + tx] = f2bf(tile[tx][ty + 8 * i]);
}

// ---------------- kernel 2: f32 -> bf16 weight convert ----------------
__global__ __launch_bounds__(256) void cvt_bf16_kernel(const float* __restrict__ in, __bf16* __restrict__ out, int n) {
    int i = blockIdx.x * 256 + threadIdx.x;
    if (i < n) out[i] = f2bf(in[i]);
}

// ---------------- kernel 3: QKV GEMM  xf[BN,256] @ w_qkv^T[256,768] -> q,k (head [M,hd]) / v (head [hd,M]) ----------------
__global__ __launch_bounds__(128) void qkv_gemm_kernel(const __bf16* __restrict__ xf,
                                                       const __bf16* __restrict__ wq,
                                                       __bf16* __restrict__ qb,
                                                       __bf16* __restrict__ kb,
                                                       __bf16* __restrict__ vb) {
    int w    = blockIdx.x * 4 + (threadIdx.x >> 5);
    int lane = threadIdx.x & 31;
    int lo = lane & 15, hi = lane >> 4;
    int ct = w % 24, rt = w / 24;             // 768/32 = 24 col tiles, 16384/32 = 512 row tiles
    int row0 = rt * 32, col0 = ct * 32;

    const __bf16* a0p = xf + (size_t)(row0 + lo) * CC;
    const __bf16* a1p = xf + (size_t)(row0 + 16 + lo) * CC;
    const __bf16* b0p = wq + (size_t)(col0 + lo) * CC;
    const __bf16* b1p = wq + (size_t)(col0 + 16 + lo) * CC;

    v8f acc00 = {}, acc01 = {}, acc10 = {}, acc11 = {};
    #pragma unroll
    for (int k0 = 0; k0 < CC; k0 += 32) {
        v16bf A0 = load_a_frag(a0p, k0, hi);
        v16bf A1 = load_a_frag(a1p, k0, hi);
        v16bf B0 = load_b_frag(b0p, k0, hi);
        v16bf B1 = load_b_frag(b1p, k0, hi);
        acc00 = WMMA_BF16(A0, B0, acc00);
        acc01 = WMMA_BF16(A0, B1, acc01);
        acc10 = WMMA_BF16(A1, B0, acc10);
        acc11 = WMMA_BF16(A1, B1, acc11);
    }

    v8f accs[2][2] = {{acc00, acc01}, {acc10, acc11}};
    #pragma unroll
    for (int a = 0; a < 2; ++a)
        #pragma unroll
        for (int bt = 0; bt < 2; ++bt)
            #pragma unroll
            for (int r = 0; r < 8; ++r) {
                int n_glob = row0 + a * 16 + r + 8 * hi;
                int c      = col0 + bt * 16 + lo;
                int b  = n_glob >> 12;            // / NN
                int nb = n_glob & (NN - 1);
                int g  = nb >> 9;                 // / MM
                int m  = nb & (MM - 1);
                int which = c >> 8;               // 0=q 1=k 2=v
                int cc2   = c & 255;
                int h = cc2 >> 5, d = cc2 & 31;
                size_t hb = (size_t)(((b * GG + g) * NH) + h) * HEADSZ;
                __bf16 val = f2bf(accs[a][bt][r]);
                if      (which == 0) qb[hb + (size_t)m * HD + d] = val;
                else if (which == 1) kb[hb + (size_t)m * HD + d] = val;
                else                 vb[hb + (size_t)d * MM + m] = val;
            }
}

// ---------------- kernel 4: flash attention, one wave per (head, 16-query tile) ----------------
__global__ __launch_bounds__(128) void attn_kernel(const __bf16* __restrict__ qb,
                                                   const __bf16* __restrict__ kb,
                                                   const __bf16* __restrict__ vb,
                                                   __bf16* __restrict__ ob) {
    int w    = blockIdx.x * 4 + (threadIdx.x >> 5);
    int lane = threadIdx.x & 31;
    int lo = lane & 15, hi = lane >> 4;
    int qt = w & 31;          // 512/16 = 32 query tiles
    int hf = w >> 5;          // flat (b,g,h) 0..255
    int h = hf & 7, g = (hf >> 3) & 7, b = hf >> 6;
    size_t hb = (size_t)hf * HEADSZ;

    const float cl2e = 0.17677669529663689f * 1.4426950408889634f; // hd^-0.5 * log2(e)

    // Q as B-fragment, loaded once: col = query (lane%16), K = hd 16*(L/16)..+15
    v16bf qf = *(const v16bf*)(qb + hb + (size_t)(qt * 16 + lo) * HD + 16 * hi);

    v8f o0 = {}, o1 = {};
    float m = -1e30f, l = 0.f;

    for (int j0 = 0; j0 < MM; j0 += 32) {
        // K rows as A-fragments (two 16-key tiles) -> S^T = K @ Q
        v16bf kA0 = load_a_frag(kb + hb + (size_t)(j0 + lo) * HD,      0, hi);
        v16bf kA1 = load_a_frag(kb + hb + (size_t)(j0 + 16 + lo) * HD, 0, hi);
        v8f z = {};
        v8f s0 = WMMA_BF16(kA0, qf, z);   // rows=keys j0+0..15,  cols=queries
        v8f s1 = WMMA_BF16(kA1, qf, z);   // rows=keys j0+16..31, cols=queries

        float t0[8], t1[8];
        float mloc = -1e30f;
        #pragma unroll
        for (int r = 0; r < 8; ++r) {
            t0[r] = s0[r] * cl2e;
            t1[r] = s1[r] * cl2e;
            mloc = fmaxf(mloc, fmaxf(t0[r], t1[r]));
        }
        mloc = fmaxf(mloc, __shfl_xor(mloc, 16, 32)); // combine key halves (same query)
        float mnew   = fmaxf(m, mloc);
        float factor = __builtin_amdgcn_exp2f(m - mnew);

        // rescale accumulators: O rows are query r+8*hi -> broadcast factor from lane (8*hi + r)
        int base = hi * 8;
        #pragma unroll
        for (int r = 0; r < 8; ++r) {
            float fr = __shfl(factor, base + r, 32);
            o0[r] *= fr;
            o1[r] *= fr;
        }

        // exponentiate; S^T C-layout *is* the P A-fragment layout (no cross-lane traffic)
        float rs = 0.f;
        v16bf pA;
        #pragma unroll
        for (int r = 0; r < 8; ++r) {
            float p0 = __builtin_amdgcn_exp2f(t0[r] - mnew);
            float p1 = __builtin_amdgcn_exp2f(t1[r] - mnew);
            rs += p0 + p1;
            pA[r]     = f2bf(p0);
            pA[r + 8] = f2bf(p1);
        }
        rs += __shfl_xor(rs, 16, 32);
        l = l * factor + rs;
        m = mnew;

        // V^T B-fragments: col = hd (lane%16 + d0), K = keys {8*hi..+7, 16+8*hi..+7} (matches P)
        v16bf vB0 = __builtin_shufflevector(*(const v8bf*)(vb + hb + (size_t)lo        * MM + j0 + 8 * hi), *(const v8bf*)(vb + hb + (size_t)lo        * MM + j0 + 16 + 8 * hi), 0,1,2,3,4,5,6,7,8,9,10,11,12,13,14,15);
        v16bf vB1 = __builtin_shufflevector(*(const v8bf*)(vb + hb + (size_t)(16 + lo) * MM + j0 + 8 * hi), *(const v8bf*)(vb + hb + (size_t)(16 + lo) * MM + j0 + 16 + 8 * hi), 0,1,2,3,4,5,6,7,8,9,10,11,12,13,14,15);
        o0 = WMMA_BF16(pA, vB0, o0);      // O = P @ V, K=32 keys in one wmma
        o1 = WMMA_BF16(pA, vB1, o1);
    }

    float linv = 1.0f / l;
    int base = hi * 8;
    #pragma unroll
    for (int r = 0; r < 8; ++r) {
        float lr = __shfl(linv, base + r, 32);
        int mrow = qt * 16 + r + 8 * hi;
        size_t oidx = ((size_t)b * NN + (size_t)g * MM + mrow) * CC + h * HD;
        ob[oidx + lo]      = f2bf(o0[r] * lr);
        ob[oidx + 16 + lo] = f2bf(o1[r] * lr);
    }
}

// ---------------- kernel 5: out-proj GEMM  at[BN,256] @ w_proj^T + b -> out [B,C,N] f32 ----------------
__global__ __launch_bounds__(128) void proj_gemm_kernel(const __bf16* __restrict__ at,
                                                        const __bf16* __restrict__ wp,
                                                        const float* __restrict__ bias,
                                                        float* __restrict__ out) {
    int w    = blockIdx.x * 4 + (threadIdx.x >> 5);
    int lane = threadIdx.x & 31;
    int lo = lane & 15, hi = lane >> 4;
    int ct = w & 7, rt = w >> 3;              // 256/32 = 8 col tiles
    int row0 = rt * 32, col0 = ct * 32;

    const __bf16* a0p = at + (size_t)(row0 + lo) * CC;
    const __bf16* a1p = at + (size_t)(row0 + 16 + lo) * CC;
    const __bf16* b0p = wp + (size_t)(col0 + lo) * CC;
    const __bf16* b1p = wp + (size_t)(col0 + 16 + lo) * CC;

    v8f acc00 = {}, acc01 = {}, acc10 = {}, acc11 = {};
    #pragma unroll
    for (int k0 = 0; k0 < CC; k0 += 32) {
        v16bf A0 = load_a_frag(a0p, k0, hi);
        v16bf A1 = load_a_frag(a1p, k0, hi);
        v16bf B0 = load_b_frag(b0p, k0, hi);
        v16bf B1 = load_b_frag(b1p, k0, hi);
        acc00 = WMMA_BF16(A0, B0, acc00);
        acc01 = WMMA_BF16(A0, B1, acc01);
        acc10 = WMMA_BF16(A1, B0, acc10);
        acc11 = WMMA_BF16(A1, B1, acc11);
    }

    v8f accs[2][2] = {{acc00, acc01}, {acc10, acc11}};
    #pragma unroll
    for (int a = 0; a < 2; ++a)
        #pragma unroll
        for (int bt = 0; bt < 2; ++bt) {
            int c  = col0 + bt * 16 + lo;
            float bv = bias[c];
            int n0 = row0 + a * 16 + 8 * hi;
            int b  = n0 >> 12;
            int nb = n0 & (NN - 1);
            v8f vals;
            #pragma unroll
            for (int r = 0; r < 8; ++r) vals[r] = accs[a][bt][r] + bv;
            // out[b][c][nb..nb+7]: 8 consecutive f32 -> one 32B store
            *(v8f*)(out + ((size_t)b * CC + c) * NN + nb) = vals;
        }
}

// ---------------- launcher ----------------
extern "C" void kernel_launch(void* const* d_in, const int* in_sizes, int n_in,
                              void* d_out, int out_size, void* d_ws, size_t ws_size,
                              hipStream_t stream) {
    const float* x      = (const float*)d_in[0];
    const float* w_qkv  = (const float*)d_in[1];
    const float* w_proj = (const float*)d_in[2];
    const float* b_proj = (const float*)d_in[3];
    float* out = (float*)d_out;

    char* ws = (char*)d_ws;
    // workspace layout (bytes)
    __bf16* xf  = (__bf16*)(ws);                 // 16384*256*2 = 8,388,608
    __bf16* wqb = (__bf16*)(ws + 8388608);       // 768*256*2   =   393,216
    __bf16* wpb = (__bf16*)(ws + 8781824);       // 256*256*2   =   131,072
    __bf16* qb  = (__bf16*)(ws + 8912896);       // 8,388,608
    __bf16* kb  = (__bf16*)(ws + 17301504);      // 8,388,608
    __bf16* vb  = (__bf16*)(ws + 25690112);      // 8,388,608
    __bf16* ab  = (__bf16*)(ws + 34078720);      // 8,388,608  (total ~40.5 MB)

    dim3 tb(32, 8, 1);
    dim3 tg(NN / 32, CC / 32, BB);
    pack_x_kernel<<<tg, tb, 0, stream>>>(x, xf);
    cvt_bf16_kernel<<<(768 * 256 + 255) / 256, 256, 0, stream>>>(w_qkv, wqb, 768 * 256);
    cvt_bf16_kernel<<<(256 * 256 + 255) / 256, 256, 0, stream>>>(w_proj, wpb, 256 * 256);

    qkv_gemm_kernel<<<(512 * 24) / 4, 128, 0, stream>>>(xf, wqb, qb, kb, vb);   // 3072 blocks
    attn_kernel<<<(256 * 32) / 4, 128, 0, stream>>>(qb, kb, vb, ab);            // 2048 blocks
    proj_gemm_kernel<<<(512 * 8) / 4, 128, 0, stream>>>(ab, wpb, b_proj, out);  // 1024 blocks
}
